// DinoV3VisionTransformer_6871947673704
// MI455X (gfx1250) — matrix operand order repaired and checked by reading, not compiled, not run on hardware
//
#include <hip/hip_runtime.h>
#include <math.h>

// ---------------------------------------------------------------------------
// DINOv3 ViT-B forward for gfx1250 (MI455X).
// - All GEMMs + both attention matmuls: v_wmma_f32_16x16x32_f16 (f32 accum)
// - Weights converted to f16 once per launch (per-layer staging buffer)
// - All GEMM inputs are f16 => LDS staging is a pure copy, done with
//   GLOBAL_LOAD_ASYNC_TO_LDS_B128 (ASYNCcnt) when the builtin exists,
//   double-buffered in the GEMM main loop.
// ---------------------------------------------------------------------------

#if defined(__has_builtin)
#  if __has_builtin(__builtin_amdgcn_global_load_async_to_lds_b128) && \
      __has_builtin(__builtin_amdgcn_s_wait_asynccnt)
#    define USE_ASYNC_LDS 1
#  endif
#endif
#ifndef USE_ASYNC_LDS
#  define USE_ASYNC_LDS 0
#endif

typedef __attribute__((ext_vector_type(16))) _Float16 v16h;
typedef __attribute__((ext_vector_type(8)))  _Float16 v8h;
typedef __attribute__((ext_vector_type(8)))  float    v8f;
typedef __attribute__((ext_vector_type(4)))  int      v4i;

union F16Frag { v16h v; v8h h[2]; };

#define NTOK   581          // 1 + 4 + 576
#define NTOKP  640          // padded to 5*128 for unguarded tiles
#define MTOK   4648         // 8 * 581
#define MPAD   4736         // 37 * 128
#define DMODEL 768
#define NHEAD  12
#define HDIM   64
#define NPATCH 576
#define HPATCH 24
#define NSPEC  5
#define DFF    3072
#define NBATCH 8
#define NLAYER 12

// 16-byte (8 half) global -> LDS copy; async on CDNA5, VGPR fallback otherwise.
__device__ __forceinline__ void cp16_async(const _Float16* g, _Float16* l) {
#if USE_ASYNC_LDS
  __builtin_amdgcn_global_load_async_to_lds_b128(
      (__attribute__((address_space(1))) v4i*)g,
      (__attribute__((address_space(3))) v4i*)l, 0, 0);
#else
  *(v8h*)l = *(const v8h*)g;
#endif
}

template<int N> __device__ __forceinline__ void wait_async() {
#if USE_ASYNC_LDS
  __builtin_amdgcn_s_wait_asynccnt(N);
#endif
}

__device__ __forceinline__ float gelu_exact(float x) {
  return 0.5f * x * (1.0f + erff(x * 0.70710678118654752440f));
}

// ---------------------------------------------------------------------------
// WMMA GEMM, all-f16 operands:  C[M,N] = A[M,K] * W[N,K]^T  (+ epilogue)
// MODE 1: outf = acc + bias*aux           (qkv: bias * bias_mask, f32 out)
// MODE 2: resid += aux * (acc + bias)     (proj / fc2 with layer-scale)
// MODE 3: outh = gelu(acc + bias)         (fc1, f16 out)
// MODE 4: outf[row-remap] = acc + bias    (patch embed -> token rows 5..580)
// 256 threads = 8 waves; tile 128x128, K-step 64, double-buffered async LDS.
// A must have >= ceil(M/128)*128 readable rows (pads zeroed by producer).
// N, K must be multiples of 128 / 64 (true for 768/2304/3072).
// ---------------------------------------------------------------------------
template<int MODE>
__global__ __launch_bounds__(256) void gemm_f16(
    const _Float16* __restrict__ A, const _Float16* __restrict__ W,
    const float* __restrict__ bias, const float* __restrict__ aux,
    float* __restrict__ outf, _Float16* __restrict__ outh,
    float* __restrict__ resid, int M, int N, int K)
{
  __shared__ _Float16 As[2][128][72];   // 64 data halves + 8 pad
  __shared__ _Float16 Bs[2][128][72];

  const int tid  = threadIdx.x;
  const int lane = tid & 31;
  const int wave = tid >> 5;
  const int wm   = wave & 1;
  const int wn   = wave >> 1;
  const int mm   = lane & 15;
  const int hi   = lane >> 4;
  const int mBase = blockIdx.y * 128;
  const int nBase = blockIdx.x * 128;

  v8f acc[4][2];
#pragma unroll
  for (int i = 0; i < 4; ++i)
#pragma unroll
    for (int j = 0; j < 2; ++j)
#pragma unroll
      for (int r = 0; r < 8; ++r) acc[i][j][r] = 0.0f;

  // copy mapping: 2 threads per row, 4 x 16B granules each
  const int srow = tid >> 1;
  const int sg   = (tid & 1) * 4;
  const _Float16* aG = A + (size_t)(mBase + srow) * K + sg * 8;
  const _Float16* bG = W + (size_t)(nBase + srow) * K + sg * 8;

  auto issue = [&](int kt, int buf) {
    const int k0 = kt << 6;
#pragma unroll
    for (int g = 0; g < 4; ++g)
      cp16_async(aG + k0 + g * 8, &As[buf][srow][(sg + g) * 8]);
#pragma unroll
    for (int g = 0; g < 4; ++g)
      cp16_async(bG + k0 + g * 8, &Bs[buf][srow][(sg + g) * 8]);
  };

  const int ksteps = K >> 6;
  issue(0, 0);
  for (int kt = 0; kt < ksteps; ++kt) {
    const int cur = kt & 1;
    if (kt + 1 < ksteps) { issue(kt + 1, cur ^ 1); wait_async<8>(); }
    else                 { wait_async<0>(); }
    __syncthreads();

#pragma unroll
    for (int ks = 0; ks < 2; ++ks) {
      const int kc = ks << 5;
      F16Frag af[4];
#pragma unroll
      for (int i = 0; i < 4; ++i) {   // A layout: k = (i/8)*16 + hi*8 + i%8
        const _Float16* rp = &As[cur][wm * 64 + i * 16 + mm][kc];
        af[i].h[0] = *(const v8h*)(rp + hi * 8);
        af[i].h[1] = *(const v8h*)(rp + 16 + hi * 8);
      }
      F16Frag bf[2];
#pragma unroll
      for (int j = 0; j < 2; ++j) {   // B layout: k = (i/8)*16 + hi*8 + i%8
        const _Float16* rp = &Bs[cur][wn * 32 + j * 16 + mm][kc];
        bf[j].h[0] = *(const v8h*)(rp + hi * 8);
        bf[j].h[1] = *(const v8h*)(rp + 16 + hi * 8);
      }
#pragma unroll
      for (int i = 0; i < 4; ++i)
#pragma unroll
        for (int j = 0; j < 2; ++j)
          acc[i][j] = __builtin_amdgcn_wmma_f32_16x16x32_f16(
              false, af[i].v, false, bf[j].v, (short)0, acc[i][j], false, false);
    }
    __syncthreads();
  }

  // epilogue: C layout -> lane col = mm, VGPR r -> row = r + 8*hi
#pragma unroll
  for (int j = 0; j < 2; ++j) {
    const int gn = nBase + wn * 32 + j * 16 + mm;
    const float bv = bias ? bias[gn] : 0.0f;
    const float av = (MODE == 1 || MODE == 2) ? aux[gn] : 0.0f;
    const float beff = (MODE == 1) ? bv * av : bv;
#pragma unroll
    for (int i = 0; i < 4; ++i) {
#pragma unroll
      for (int r = 0; r < 8; ++r) {
        const int gm = mBase + wm * 64 + i * 16 + hi * 8 + r;
        if (gm >= M) continue;
        float v = acc[i][j][r] + beff;
        if (MODE == 3) {
          outh[(size_t)gm * N + gn] = (_Float16)gelu_exact(v);
        } else if (MODE == 2) {
          resid[(size_t)gm * N + gn] += av * v;
        } else if (MODE == 4) {
          const int bb = gm / NPATCH, pp = gm % NPATCH;
          outf[((size_t)(bb * NTOK + NSPEC + pp)) * N + gn] = v;
        } else {
          outf[(size_t)gm * N + gn] = v;
        }
      }
    }
  }
}

// ---------------------------------------------------------------------------
// LayerNorm: one wave per 768-wide row; OUT16 selects f16 vs f32 output
// ---------------------------------------------------------------------------
template<int OUT16>
__global__ __launch_bounds__(256) void ln_kernel(
    const float* __restrict__ x, const float* __restrict__ s,
    const float* __restrict__ b, float* __restrict__ outf,
    _Float16* __restrict__ outh, int rows)
{
  const int wave = threadIdx.x >> 5;
  const int lane = threadIdx.x & 31;
  const int row = blockIdx.x * 8 + wave;
  if (row >= rows) return;
  const float* xr = x + (size_t)row * DMODEL;
  float v[24];
  float sum = 0.0f;
#pragma unroll
  for (int j = 0; j < 24; ++j) { v[j] = xr[lane + (j << 5)]; sum += v[j]; }
#pragma unroll
  for (int off = 16; off >= 1; off >>= 1) sum += __shfl_xor(sum, off, 32);
  const float mu = sum * (1.0f / 768.0f);
  float vs = 0.0f;
#pragma unroll
  for (int j = 0; j < 24; ++j) { const float d = v[j] - mu; vs += d * d; }
#pragma unroll
  for (int off = 16; off >= 1; off >>= 1) vs += __shfl_xor(vs, off, 32);
  const float rinv = rsqrtf(vs * (1.0f / 768.0f) + 1e-6f);
#pragma unroll
  for (int j = 0; j < 24; ++j) {
    const int c = lane + (j << 5);
    const float o = (v[j] - mu) * rinv * s[c] + b[c];
    if (OUT16) outh[(size_t)row * DMODEL + c] = (_Float16)o;
    else       outf[(size_t)row * DMODEL + c] = o;
  }
}

// ---------------------------------------------------------------------------
// RoPE split -> f16: Q,K [bh, 640(pad), 64] rotated; Vt [bh, 64, 640(pad)]
// ---------------------------------------------------------------------------
__global__ __launch_bounds__(64) void rope_kernel(
    const float* __restrict__ qkv, const float* __restrict__ periods,
    _Float16* __restrict__ Q, _Float16* __restrict__ Kout,
    _Float16* __restrict__ Vt)
{
  const int d = threadIdx.x;                 // 0..63
  const int idx = blockIdx.x;                // bh * NTOK + t
  const int t = idx % NTOK;
  const int bh = idx / NTOK;
  const int hh = bh % NHEAD;
  const int b = bh / NHEAD;
  const size_t base = ((size_t)(b * NTOK + t)) * (3 * DMODEL);
  const float* qs = qkv + base + hh * HDIM;
  const float* ks = qkv + base + DMODEL + hh * HDIM;
  const float* vsrc = qkv + base + 2 * DMODEL + hh * HDIM;
  const size_t obase = ((size_t)bh * NTOKP + t) * HDIM;

  Vt[((size_t)bh * HDIM + d) * NTOKP + t] = (_Float16)vsrc[d];

  float qv, kv;
  if (t < NSPEC) {
    qv = qs[d]; kv = ks[d];
  } else {
    const int p = t - NSPEC;
    const int py = p / HPATCH;
    const int px = p % HPATCH;
    const int half = d >> 5;                 // 0: x-rope, 1: y-rope
    const int f = (d & 31) >> 1;             // freq index 0..15
    const float freq = 6.283185307179586f / periods[f];
    const float ang = (half ? (float)py : (float)px) * freq;
    const float c = cosf(ang);
    const float sn = sinf(ang);
    const int pb = (half << 5) + (f << 1);
    const float qe = qs[pb], qo = qs[pb + 1];
    const float ke = ks[pb], ko = ks[pb + 1];
    if (d & 1) { qv = qe * sn + qo * c; kv = ke * sn + ko * c; }
    else       { qv = qe * c - qo * sn; kv = ke * c - ko * sn; }
  }
  Q[obase + d] = (_Float16)qv;
  Kout[obase + d] = (_Float16)kv;
}

// ---------------------------------------------------------------------------
// Flash-style attention (f16 in, f16 out), async LDS staging of K / V tiles.
// One (b,h,q-tile 128) per block, 8 waves; online softmax with shfl reduces.
// q/k padded to 640 rows per (b,h), vt padded to 640 cols (pads are zero).
// ---------------------------------------------------------------------------
__global__ __launch_bounds__(256) void attn_kernel(
    const _Float16* __restrict__ Q, const _Float16* __restrict__ Kmat,
    const _Float16* __restrict__ Vt, _Float16* __restrict__ Out)
{
  __shared__ _Float16 Ks[128][72];
  __shared__ _Float16 Vs[64][136];
  __shared__ _Float16 Ps[8][16][40];   // per-wave 16x32 P chunk (+pad)

  const int tid = threadIdx.x;
  const int lane = tid & 31;
  const int wave = tid >> 5;
  const int mm = lane & 15;
  const int hi = lane >> 4;
  const int qt = blockIdx.x;
  const int bh = blockIdx.y;
  const int b = bh / NHEAD;
  const int hh = bh % NHEAD;
  const int qbase = qt * 128;

  const _Float16* Qb = Q    + (size_t)bh * NTOKP * HDIM;
  const _Float16* Kb = Kmat + (size_t)bh * NTOKP * HDIM;
  const _Float16* Vb = Vt   + (size_t)bh * HDIM * NTOKP;

  // Q fragments (A operand) straight from global f16 (padded rows are zero)
  const int qrow = qbase + wave * 16 + mm;
  const _Float16* qp = Qb + (size_t)qrow * HDIM;
  F16Frag aq[2];
#pragma unroll
  for (int ks = 0; ks < 2; ++ks) {
    aq[ks].h[0] = *(const v8h*)(qp + ks * 32 + hi * 8);
    aq[ks].h[1] = *(const v8h*)(qp + ks * 32 + 16 + hi * 8);
  }

  float mrun[8], lrun[8];
  v8f oacc[4];
#pragma unroll
  for (int r = 0; r < 8; ++r) { mrun[r] = -3.0e38f; lrun[r] = 0.0f; }
#pragma unroll
  for (int jd = 0; jd < 4; ++jd)
#pragma unroll
    for (int r = 0; r < 8; ++r) oacc[jd][r] = 0.0f;

  // async copy mappings
  const int kr = tid >> 1, kg = (tid & 1) * 4;   // K: 2 thr/row, 4 granules
  const int vr = tid >> 2, vg = (tid & 3) * 4;   // V: 4 thr/row, 4 granules

  for (int kt = 0; kt < 5; ++kt) {
    const int nb = kt * 128;
#pragma unroll
    for (int g = 0; g < 4; ++g)
      cp16_async(Kb + (size_t)(nb + kr) * HDIM + (kg + g) * 8,
                 &Ks[kr][(kg + g) * 8]);
#pragma unroll
    for (int g = 0; g < 4; ++g)
      cp16_async(Vb + (size_t)vr * NTOKP + nb + (vg + g) * 8,
                 &Vs[vr][(vg + g) * 8]);
    wait_async<0>();
    __syncthreads();

    // S = Q K^T (16 wmma per wave)
    v8f sacc[8];
#pragma unroll
    for (int j = 0; j < 8; ++j)
#pragma unroll
      for (int r = 0; r < 8; ++r) sacc[j][r] = 0.0f;
#pragma unroll
    for (int ks = 0; ks < 2; ++ks) {
#pragma unroll
      for (int j = 0; j < 8; ++j) {
        F16Frag bk;
        const _Float16* rp = &Ks[j * 16 + mm][ks * 32];
        bk.h[0] = *(const v8h*)(rp + hi * 8);
        bk.h[1] = *(const v8h*)(rp + 16 + hi * 8);
        sacc[j] = __builtin_amdgcn_wmma_f32_16x16x32_f16(
            false, aq[ks].v, false, bk.v, (short)0, sacc[j], false, false);
      }
    }

    // scale + mask (column = nb + j*16 + mm)
#pragma unroll
    for (int j = 0; j < 8; ++j) {
      const bool oob = (nb + j * 16 + mm) >= NTOK;
#pragma unroll
      for (int r = 0; r < 8; ++r)
        sacc[j][r] = oob ? -3.0e38f : sacc[j][r] * 0.125f;
    }

    // online softmax (rows live in 16-lane halves -> xor masks 8..1)
#pragma unroll
    for (int r = 0; r < 8; ++r) {
      float mx = sacc[0][r];
#pragma unroll
      for (int j = 1; j < 8; ++j) mx = fmaxf(mx, sacc[j][r]);
#pragma unroll
      for (int off = 8; off >= 1; off >>= 1) mx = fmaxf(mx, __shfl_xor(mx, off, 32));
      const float mnew = fmaxf(mrun[r], mx);
      const float corr = __expf(mrun[r] - mnew);
      mrun[r] = mnew;
      float rs = 0.0f;
#pragma unroll
      for (int j = 0; j < 8; ++j) {
        const float pv = __expf(sacc[j][r] - mnew);
        sacc[j][r] = pv;
        rs += pv;
      }
#pragma unroll
      for (int off = 8; off >= 1; off >>= 1) rs += __shfl_xor(rs, off, 32);
      lrun[r] = lrun[r] * corr + rs;
#pragma unroll
      for (int jd = 0; jd < 4; ++jd) oacc[jd][r] *= corr;
    }

    // O += P * V  (per 32-col chunk: C->A layout via per-wave LDS buffer)
#pragma unroll
    for (int ks2 = 0; ks2 < 4; ++ks2) {
#pragma unroll
      for (int jj = 0; jj < 2; ++jj) {
        const int j = ks2 * 2 + jj;
#pragma unroll
        for (int r = 0; r < 8; ++r)
          Ps[wave][hi * 8 + r][jj * 16 + mm] = (_Float16)sacc[j][r];
      }
      F16Frag ap;
      const _Float16* pr = &Ps[wave][mm][0];
      ap.h[0] = *(const v8h*)(pr + hi * 8);
      ap.h[1] = *(const v8h*)(pr + 16 + hi * 8);
#pragma unroll
      for (int jd = 0; jd < 4; ++jd) {
        F16Frag bv;
        const _Float16* rp = &Vs[jd * 16 + mm][ks2 * 32];
        bv.h[0] = *(const v8h*)(rp + hi * 8);
        bv.h[1] = *(const v8h*)(rp + 16 + hi * 8);
        oacc[jd] = __builtin_amdgcn_wmma_f32_16x16x32_f16(
            false, ap.v, false, bv.v, (short)0, oacc[jd], false, false);
      }
    }
    __syncthreads();
  }

  // normalize + write f16 into proj-input layout [MPAD, 768]
#pragma unroll
  for (int jd = 0; jd < 4; ++jd) {
#pragma unroll
    for (int r = 0; r < 8; ++r) {
      const int row = qbase + wave * 16 + hi * 8 + r;
      if (row < NTOK) {
        const int col = hh * HDIM + jd * 16 + mm;
        Out[((size_t)(b * NTOK + row)) * DMODEL + col] =
            (_Float16)(oacc[jd][r] / lrun[r]);
      }
    }
  }
}

// ---------------------------------------------------------------------------
// im2col (f16 out): out[m,k], m = b*576 + hp*24 + wp, k = c*256 + py*16 + px
// ---------------------------------------------------------------------------
__global__ __launch_bounds__(256) void im2col_kernel(
    const float* __restrict__ px, _Float16* __restrict__ out)
{
  const long long idx = (long long)blockIdx.x * 256 + threadIdx.x;
  const long long total = (long long)NBATCH * NPATCH * DMODEL;
  if (idx >= total) return;
  const int m = (int)(idx / DMODEL);
  const int k = (int)(idx % DMODEL);
  const int b = m / NPATCH;
  const int p = m % NPATCH;
  const int hp = p / HPATCH, wp = p % HPATCH;
  const int c = k >> 8;
  const int rem = k & 255;
  const int py = rem >> 4, pxx = rem & 15;
  out[idx] = (_Float16)px[(((size_t)b * 3 + c) * 384 + hp * 16 + py) * 384
                          + wp * 16 + pxx];
}

// cls + storage token rows of the residual stream (f32)
__global__ __launch_bounds__(256) void assemble_kernel(
    const float* __restrict__ cls, const float* __restrict__ storage,
    float* __restrict__ h)
{
  const int idx = blockIdx.x * 256 + threadIdx.x;
  const int total = NBATCH * NSPEC * DMODEL;
  if (idx >= total) return;
  const int b = idx / (NSPEC * DMODEL);
  const int r = (idx / DMODEL) % NSPEC;
  const int d = idx % DMODEL;
  h[((size_t)(b * NTOK + r)) * DMODEL + d] =
      (r == 0) ? cls[d] : storage[(r - 1) * DMODEL + d];
}

// f32 -> f16 weight conversion (n multiple of 8)
__global__ __launch_bounds__(256) void cvt16_kernel(
    const float* __restrict__ src, _Float16* __restrict__ dst, long long n)
{
  const long long i = ((long long)blockIdx.x * 256 + threadIdx.x) * 8;
  if (i >= n) return;
  const float4 a = *(const float4*)(src + i);
  const float4 c = *(const float4*)(src + i + 4);
  v8h r;
  r[0] = (_Float16)a.x; r[1] = (_Float16)a.y; r[2] = (_Float16)a.z; r[3] = (_Float16)a.w;
  r[4] = (_Float16)c.x; r[5] = (_Float16)c.y; r[6] = (_Float16)c.z; r[7] = (_Float16)c.w;
  *(v8h*)(dst + i) = r;
}

// zero a f16 region (n multiple of 16) — establishes pad rows/cols each call
__global__ __launch_bounds__(256) void zero16_kernel(
    _Float16* __restrict__ p, long long n)
{
  const long long i = ((long long)blockIdx.x * 256 + threadIdx.x) * 16;
  if (i >= n) return;
  v8h z;
#pragma unroll
  for (int j = 0; j < 8; ++j) z[j] = (_Float16)0.0f;
  *(v8h*)(p + i) = z;
  *(v8h*)(p + i + 8) = z;
}

// ---------------------------------------------------------------------------
extern "C" void kernel_launch(void* const* d_in, const int* in_sizes, int n_in,
                              void* d_out, int out_size, void* d_ws, size_t ws_size,
                              hipStream_t stream) {
  (void)in_sizes; (void)n_in; (void)out_size;

  const float* pixel     = (const float*)d_in[0];
  const float* conv_w    = (const float*)d_in[1];   // [768, 768]
  const float* conv_b    = (const float*)d_in[2];
  const float* cls       = (const float*)d_in[3];
  const float* storage   = (const float*)d_in[4];
  const float* periods   = (const float*)d_in[5];
  const float* ln1_s     = (const float*)d_in[6];
  const float* ln1_b     = (const float*)d_in[7];
  const float* qkv_w     = (const float*)d_in[8];   // [12, 2304, 768]
  const float* qkv_b     = (const float*)d_in[9];
  const float* bias_mask = (const float*)d_in[10];
  const float* proj_w    = (const float*)d_in[11];  // [12, 768, 768]
  const float* proj_b    = (const float*)d_in[12];
  const float* ls1       = (const float*)d_in[13];
  const float* ln2_s     = (const float*)d_in[14];
  const float* ln2_b     = (const float*)d_in[15];
  const float* fc1_w     = (const float*)d_in[16];  // [12, 3072, 768]
  const float* fc1_b     = (const float*)d_in[17];
  const float* fc2_w     = (const float*)d_in[18];  // [12, 768, 3072]
  const float* fc2_b     = (const float*)d_in[19];
  const float* ls2       = (const float*)d_in[20];
  const float* lnf_s     = (const float*)d_in[21];
  const float* lnf_b     = (const float*)d_in[22];

  // f16 region sizes (halves)
  const size_t H_XLN = (size_t)MPAD * DMODEL;           // 3,637,248
  const size_t H_ATN = H_XLN;
  const size_t H_MLP = (size_t)MPAD * DFF;              // 14,548,992
  const size_t H_QK  = (size_t)NBATCH * NHEAD * NTOKP * HDIM;  // 3,932,160
  const size_t H_IM  = (size_t)NBATCH * NPATCH * DMODEL;
  const size_t HW_QKV  = (size_t)3 * DMODEL * DMODEL;
  const size_t HW_PROJ = (size_t)DMODEL * DMODEL;
  const size_t HW_FC1  = (size_t)DFF * DMODEL;
  const size_t HW_FC2  = (size_t)DMODEL * DFF;
  const size_t HW_CONV = (size_t)DMODEL * DMODEL;
  // f32 sizes (floats)
  const size_t SZ_MD  = (size_t)MTOK * DMODEL;
  const size_t SZ_QKV = (size_t)MTOK * 3 * DMODEL;

  const size_t ZERO_N = H_XLN + H_ATN + H_MLP + 3 * H_QK;   // contiguous
  const size_t need =
      (ZERO_N + H_IM + HW_QKV + HW_PROJ + HW_FC1 + HW_FC2 + HW_CONV) * 2 +
      (SZ_MD + SZ_QKV) * 4;
  if (ws_size < need) return;

  _Float16* xln16 = (_Float16*)d_ws;
  _Float16* atn16 = xln16 + H_XLN;
  _Float16* mlp16 = atn16 + H_ATN;
  _Float16* q16   = mlp16 + H_MLP;
  _Float16* k16   = q16 + H_QK;
  _Float16* vt16  = k16 + H_QK;
  _Float16* im16  = vt16 + H_QK;
  _Float16* wq16  = im16 + H_IM;
  _Float16* wp16  = wq16 + HW_QKV;
  _Float16* w116  = wp16 + HW_PROJ;
  _Float16* w216  = w116 + HW_FC1;
  _Float16* wc16  = w216 + HW_FC2;
  float* h    = (float*)(wc16 + HW_CONV);
  float* qkvb = h + SZ_MD;

  const dim3 blk(256);
  const int MT = MPAD / 128;                 // 37
  const int MP = (NBATCH * NPATCH) / 128;    // 36

  // establish zero pads in all padded f16 activation buffers
  zero16_kernel<<<(unsigned)(ZERO_N / (256 * 16)), blk, 0, stream>>>(
      xln16, (long long)ZERO_N);

  // patch embed: im2col + conv-as-GEMM -> token rows 5..580 of h
  cvt16_kernel<<<(unsigned)(HW_CONV / 2048), blk, 0, stream>>>(conv_w, wc16, (long long)HW_CONV);
  im2col_kernel<<<(unsigned)((H_IM + 255) / 256), blk, 0, stream>>>(pixel, im16);
  gemm_f16<4><<<dim3(DMODEL / 128, MP), blk, 0, stream>>>(
      im16, wc16, conv_b, nullptr, h, nullptr, nullptr,
      NBATCH * NPATCH, DMODEL, DMODEL);
  assemble_kernel<<<(NBATCH * NSPEC * DMODEL + 255) / 256, blk, 0, stream>>>(cls, storage, h);

  for (int l = 0; l < NLAYER; ++l) {
    cvt16_kernel<<<(unsigned)(HW_QKV  / 2048), blk, 0, stream>>>(qkv_w  + (size_t)l * HW_QKV,  wq16, (long long)HW_QKV);
    cvt16_kernel<<<(unsigned)(HW_PROJ / 2048), blk, 0, stream>>>(proj_w + (size_t)l * HW_PROJ, wp16, (long long)HW_PROJ);
    cvt16_kernel<<<(unsigned)(HW_FC1  / 2048), blk, 0, stream>>>(fc1_w  + (size_t)l * HW_FC1,  w116, (long long)HW_FC1);
    cvt16_kernel<<<(unsigned)(HW_FC2  / 2048), blk, 0, stream>>>(fc2_w  + (size_t)l * HW_FC2,  w216, (long long)HW_FC2);

    ln_kernel<1><<<MTOK / 8, blk, 0, stream>>>(
        h, ln1_s + l * DMODEL, ln1_b + l * DMODEL, nullptr, xln16, MTOK);
    gemm_f16<1><<<dim3((3 * DMODEL) / 128, MT), blk, 0, stream>>>(
        xln16, wq16, qkv_b + l * 3 * DMODEL, bias_mask + l * 3 * DMODEL,
        qkvb, nullptr, nullptr, MTOK, 3 * DMODEL, DMODEL);
    rope_kernel<<<NBATCH * NHEAD * NTOK, dim3(64), 0, stream>>>(
        qkvb, periods, q16, k16, vt16);
    attn_kernel<<<dim3(5, NBATCH * NHEAD), blk, 0, stream>>>(q16, k16, vt16, atn16);
    gemm_f16<2><<<dim3(DMODEL / 128, MT), blk, 0, stream>>>(
        atn16, wp16, proj_b + l * DMODEL, ls1 + l * DMODEL,
        nullptr, nullptr, h, MTOK, DMODEL, DMODEL);
    ln_kernel<1><<<MTOK / 8, blk, 0, stream>>>(
        h, ln2_s + l * DMODEL, ln2_b + l * DMODEL, nullptr, xln16, MTOK);
    gemm_f16<3><<<dim3(DFF / 128, MT), blk, 0, stream>>>(
        xln16, w116, fc1_b + l * DFF, nullptr,
        nullptr, mlp16, nullptr, MTOK, DFF, DMODEL);
    gemm_f16<2><<<dim3(DMODEL / 128, MT), blk, 0, stream>>>(
        mlp16, w216, fc2_b + l * DMODEL, ls2 + l * DMODEL,
        nullptr, nullptr, h, MTOK, DMODEL, DFF);
  }

  ln_kernel<0><<<MTOK / 8, blk, 0, stream>>>(
      h, lnf_s, lnf_b, (float*)d_out, nullptr, MTOK);
}
